// LSTM_28054726377935
// MI455X (gfx1250) — matrix operand order repaired and checked by reading, not compiled, not run on hardware
//
#include <hip/hip_runtime.h>
#include <hip/hip_bf16.h>

// ---------------- problem constants ----------------
#define TT   512
#define BB   64
#define II   1024
#define HH   1024
#define NG   4096            // 4*H
#define K2   2048            // I + H
#define NWG  64              // persistent workgroups (N tiles of 64 cols)

// ---------------- workspace layout (bytes) ----------------
// wfrag : [256 ntile][64 ktile][32 lane][16 bf16]  = 16 MB
// xfrag : [512 t][4 mtile][32 ktile][32 lane][16]  = 64 MB
// hfrag : [4 mtile][32 ktile][32 lane][16]         = 128 KB
// gates : [64][4096] f32                           = 1 MB
// cbuf  : [64][1024] f32                           = 256 KB
// bar   : 16 u32
#define WFRAG_OFF  0UL
#define XFRAG_OFF  (16UL*1024*1024)
#define HFRAG_OFF  (XFRAG_OFF + 64UL*1024*1024)
#define GATES_OFF  (HFRAG_OFF + 128UL*1024)
#define CBUF_OFF   (GATES_OFF + 1UL*1024*1024)
#define BAR_OFF    (CBUF_OFF + 256UL*1024)

typedef __attribute__((ext_vector_type(16))) __bf16 v16bf;
typedef __attribute__((ext_vector_type(8)))  float  v8f;

__device__ __forceinline__ unsigned short f32_to_bf16_bits(float f) {
    union { float f; unsigned int u; } v; v.f = f;
    unsigned int u = v.u;
    u += 0x7FFFu + ((u >> 16) & 1u);      // round-to-nearest-even
    return (unsigned short)(u >> 16);
}

__device__ __forceinline__ float sigmoid_f(float x) {
    return 1.0f / (1.0f + __expf(-x));
}
__device__ __forceinline__ float tanh_f(float x) {
    x = fminf(15.0f, fmaxf(-15.0f, x));
    float e = __expf(2.0f * x);
    return (e - 1.0f) / (e + 1.0f);
}

__device__ __forceinline__ v8f wmma_bf16(v16bf a, v16bf b, v8f c) {
    // (neg_a, A, neg_b, B, c_mod, C, reuse_a, reuse_b)
    return __builtin_amdgcn_wmma_f32_16x16x32_bf16(false, a, false, b, (short)0, c, false, false);
}

// ---------------------------------------------------------------------------
// Pack W (gate order f,i,C,o along N; rows 0..1023 = h part, 1024..2047 = x
// part, matching xh=[h,x]) into B-matrix wave32 fragments:
//   col = lane&15 ; K = ktile*32 + (lane>=16 ? 8:0) + (j<8 ? j : j+8)
// ---------------------------------------------------------------------------
__global__ void lstm_pack_w_kernel(const float* __restrict__ Wf,
                                   const float* __restrict__ Wi,
                                   const float* __restrict__ WC,
                                   const float* __restrict__ Wo,
                                   unsigned short* __restrict__ wfrag) {
    unsigned int idx = blockIdx.x * 256u + threadIdx.x;   // 8,388,608 total
    unsigned int j    =  idx        & 15u;
    unsigned int lane = (idx >> 4)  & 31u;
    unsigned int kt   = (idx >> 9)  & 63u;
    unsigned int nt   =  idx >> 15;
    unsigned int n    = nt * 16u + (lane & 15u);
    unsigned int gate = n >> 10;
    unsigned int ncol = n & 1023u;
    unsigned int k    = kt * 32u + ((lane >> 4) << 3) + ((j < 8u) ? j : j + 8u);
    const float* Wg = (gate == 0u) ? Wf : (gate == 1u) ? Wi : (gate == 2u) ? WC : Wo;
    wfrag[idx] = f32_to_bf16_bits(Wg[k * 1024u + ncol]);
}

// ---------------------------------------------------------------------------
// Pack x[t][b][i] into A-matrix wave32 fragments (16-bit A 16x32 layout):
//   row = lane&15 ; K = ktile*32 + (lane>=16 ? 8:0) + (j<8 ? j : j+8)
// ---------------------------------------------------------------------------
__global__ void lstm_pack_x_kernel(const float* __restrict__ x,
                                   unsigned short* __restrict__ xfrag) {
    unsigned int idx = blockIdx.x * 256u + threadIdx.x;   // 33,554,432 total
    unsigned int j    =  idx        & 15u;
    unsigned int lane = (idx >> 4)  & 31u;
    unsigned int kt   = (idx >> 9)  & 31u;
    unsigned int mt   = (idx >> 14) & 3u;
    unsigned int t    =  idx >> 16;
    unsigned int b    = mt * 16u + (lane & 15u);
    unsigned int k    = kt * 32u + ((lane >> 4) << 3) + ((j < 8u) ? j : j + 8u);
    xfrag[idx] = f32_to_bf16_bits(x[(t * 64u + b) * 1024u + k]);
}

// ---------------------------------------------------------------------------
// Init h fragments from h0, c buffer from c0, reset barrier counters.
// ---------------------------------------------------------------------------
__global__ void lstm_init_kernel(const float* __restrict__ h0,
                                 const float* __restrict__ c0,
                                 unsigned short* __restrict__ hfrag,
                                 float* __restrict__ cbuf,
                                 unsigned int* __restrict__ bar) {
    unsigned int idx = blockIdx.x * 256u + threadIdx.x;   // 65,536 total
    unsigned int j    =  idx        & 15u;
    unsigned int lane = (idx >> 4)  & 31u;
    unsigned int kt   = (idx >> 9)  & 31u;
    unsigned int mt   = (idx >> 14) & 3u;
    unsigned int b    = mt * 16u + (lane & 15u);
    unsigned int k    = kt * 32u + ((lane >> 4) << 3) + ((j < 8u) ? j : j + 8u);
    hfrag[idx] = f32_to_bf16_bits(h0[b * 1024u + k]);
    cbuf[idx]  = c0[idx];
    if (idx < 16u) bar[idx] = 0u;
}

// ---------------------------------------------------------------------------
// Grid-wide barrier for the persistent kernel (all NWG workgroups resident).
// ---------------------------------------------------------------------------
__device__ __forceinline__ void grid_barrier(unsigned int* bar, unsigned int gen) {
    __syncthreads();
    if (threadIdx.x == 0) {
        __threadfence();
        unsigned int prev = __hip_atomic_fetch_add(&bar[0], 1u, __ATOMIC_ACQ_REL,
                                                   __HIP_MEMORY_SCOPE_AGENT);
        if (prev == (unsigned int)(NWG - 1)) {
            __hip_atomic_store(&bar[0], 0u, __ATOMIC_RELAXED, __HIP_MEMORY_SCOPE_AGENT);
            __hip_atomic_fetch_add(&bar[1], 1u, __ATOMIC_RELEASE, __HIP_MEMORY_SCOPE_AGENT);
        } else {
            while (__hip_atomic_load(&bar[1], __ATOMIC_ACQUIRE,
                                     __HIP_MEMORY_SCOPE_AGENT) <= gen) {
                __builtin_amdgcn_s_sleep(2);
            }
        }
    }
    __syncthreads();
}

// ---------------------------------------------------------------------------
// Persistent LSTM: 64 WGs x 256 threads. Per step:
//  phase 1: gates[64,4096] = [h,x_t] @ W   (bf16 WMMA, f32 accum, L2-resident)
//  phase 2: elementwise gate math, update c/h, write h_seq + next h fragments
// ---------------------------------------------------------------------------
__global__ void lstm_persistent_kernel(const unsigned short* __restrict__ wfrag,
                                       const unsigned short* __restrict__ xfrag,
                                       unsigned short* __restrict__ hfrag,
                                       float* __restrict__ gates,
                                       float* __restrict__ cbuf,
                                       unsigned int* __restrict__ bar,
                                       const float* __restrict__ b_f,
                                       const float* __restrict__ b_i,
                                       const float* __restrict__ b_C,
                                       const float* __restrict__ b_o,
                                       float* __restrict__ out) {
    const int wg   = blockIdx.x;
    const int tid  = threadIdx.x;
    const int lane = tid & 31;
    const int wave = tid >> 5;
    const int m_w  = wave & 3;        // M tile (16 rows each, M=64)
    const int n_w  = wave >> 2;       // 0..1
    const int nt0  = wg * 4 + n_w * 2; // first of two 16-col N tiles

    const v16bf* wb0 = (const v16bf*)wfrag + ((size_t)nt0 * 64u * 32u + (unsigned)lane);
    const v16bf* wb1 = wb0 + 64u * 32u;
    const v16bf* ha  = (const v16bf*)hfrag + ((size_t)m_w * 32u * 32u + (unsigned)lane);

    float* out_hseq  = out;
    float* out_hlast = out + (size_t)TT * BB * HH;
    float* out_clast = out_hlast + (size_t)BB * HH;

    unsigned int gen = 0;

    for (int t = 0; t < TT; ++t) {
        // ---------------- phase 1: fused-gate GEMM ----------------
        const v16bf* xa = (const v16bf*)xfrag +
                          (((size_t)t * 4u + (unsigned)m_w) * 32u * 32u + (unsigned)lane);
        v8f acc0 = {};
        v8f acc1 = {};

        // h part: K tiles 0..31
        #pragma unroll 4
        for (int kt = 0; kt < 32; ++kt) {
            v16bf a  = ha[(size_t)kt * 32u];
            v16bf bA = wb0[(size_t)kt * 32u];
            v16bf bB = wb1[(size_t)kt * 32u];
            __builtin_prefetch(&wb0[(size_t)(kt + 4) * 32u], 0, 1);
            __builtin_prefetch(&wb1[(size_t)(kt + 4) * 32u], 0, 1);
            acc0 = wmma_bf16(a, bA, acc0);
            acc1 = wmma_bf16(a, bB, acc1);
        }
        // x part: K tiles 32..63 (weight rows 1024..2047)
        #pragma unroll 4
        for (int kt = 0; kt < 32; ++kt) {
            v16bf a  = xa[(size_t)kt * 32u];
            v16bf bA = wb0[(size_t)(32 + kt) * 32u];
            v16bf bB = wb1[(size_t)(32 + kt) * 32u];
            __builtin_prefetch(&xa[(size_t)(kt + 4) * 32u], 0, 1);
            acc0 = wmma_bf16(a, bA, acc0);
            acc1 = wmma_bf16(a, bB, acc1);
        }

        // store D: VGPR r -> M = m_w*16 + (lane>=16 ? 8:0) + r ; N col = lane&15
        {
            const int mbase = m_w * 16 + ((lane >> 4) << 3);
            const int ncol  = lane & 15;
            #pragma unroll
            for (int r = 0; r < 8; ++r) {
                gates[(size_t)(mbase + r) * NG + nt0 * 16 + ncol]       = acc0[r];
                gates[(size_t)(mbase + r) * NG + (nt0 + 1) * 16 + ncol] = acc1[r];
            }
        }

        grid_barrier(bar, gen); ++gen;

        // ---------------- phase 2: elementwise LSTM cell ----------------
        const int gtid = wg * 256 + tid;              // 16384 threads, 65536 cells
        #pragma unroll
        for (int rep = 0; rep < 4; ++rep) {
            int idx = gtid + rep * (NWG * 256);
            int b_row = idx >> 10;
            int n     = idx & 1023;
            const float* grow = gates + (size_t)b_row * NG;
            float gf = grow[n]          + b_f[n];
            float gi = grow[1024 + n]   + b_i[n];
            float gg = grow[2048 + n]   + b_C[n];
            float go = grow[3072 + n]   + b_o[n];
            float f_t = sigmoid_f(gf);
            float i_t = sigmoid_f(gi);
            float g_t = tanh_f(gg);
            float o_t = sigmoid_f(go);
            float c_t = f_t * cbuf[idx] + i_t * g_t;
            float h_t = o_t * tanh_f(c_t);
            cbuf[idx] = c_t;
            out_hseq[(size_t)t * (BB * HH) + idx] = h_t;
            if (t == TT - 1) {
                out_hlast[idx] = h_t;
                out_clast[idx] = c_t;
            }
            // scatter h into A-fragment layout for next step's GEMM
            int mt   = b_row >> 4;
            int lrow = b_row & 15;
            int ktl  = n >> 5;
            int kin  = n & 31;
            int lane2 = lrow + (((kin >> 3) & 1) << 4);
            int j2    = (kin & 7) + ((kin & 16) ? 8 : 0);
            hfrag[((((size_t)mt * 32u + (unsigned)ktl) * 32u + (unsigned)lane2) << 4) + j2] =
                f32_to_bf16_bits(h_t);
        }

        grid_barrier(bar, gen); ++gen;
    }
}

// ---------------------------------------------------------------------------
extern "C" void kernel_launch(void* const* d_in, const int* in_sizes, int n_in,
                              void* d_out, int out_size, void* d_ws, size_t ws_size,
                              hipStream_t stream) {
    const float* x  = (const float*)d_in[0];
    const float* h0 = (const float*)d_in[1];
    const float* c0 = (const float*)d_in[2];
    const float* Wf = (const float*)d_in[3];
    const float* bf = (const float*)d_in[4];
    const float* Wi = (const float*)d_in[5];
    const float* bi = (const float*)d_in[6];
    const float* WC = (const float*)d_in[7];
    const float* bC = (const float*)d_in[8];
    const float* Wo = (const float*)d_in[9];
    const float* bo = (const float*)d_in[10];
    float* out = (float*)d_out;

    char* ws = (char*)d_ws;
    unsigned short* wfrag = (unsigned short*)(ws + WFRAG_OFF);
    unsigned short* xfrag = (unsigned short*)(ws + XFRAG_OFF);
    unsigned short* hfrag = (unsigned short*)(ws + HFRAG_OFF);
    float*          gates = (float*)(ws + GATES_OFF);
    float*          cbuf  = (float*)(ws + CBUF_OFF);
    unsigned int*   bar   = (unsigned int*)(ws + BAR_OFF);

    lstm_pack_w_kernel<<<dim3(32768), dim3(256), 0, stream>>>(Wf, Wi, WC, Wo, wfrag);
    lstm_pack_x_kernel<<<dim3(131072), dim3(256), 0, stream>>>(x, xfrag);
    lstm_init_kernel<<<dim3(256), dim3(256), 0, stream>>>(h0, c0, hfrag, cbuf, bar);
    lstm_persistent_kernel<<<dim3(NWG), dim3(256), 0, stream>>>(
        wfrag, xfrag, hfrag, gates, cbuf, bar, bf, bi, bC, bo, out);
}
